// DUST_65085934403760
// MI455X (gfx1250) — matrix operand order repaired and checked by reading, not compiled, not run on hardware
//
#include <hip/hip_runtime.h>

#define W_DIM   1024
#define D_DIM   2048
#define BATCH_N 2048
#define P_N     8
#define OMEGA_K 50
#define NITERS  10
#define CLAMP_V 150.0f

typedef __attribute__((ext_vector_type(16))) __bf16 v16bf;
typedef __attribute__((ext_vector_type(8)))  float  v8f;
typedef int v4i_t __attribute__((vector_size(16)));   // int4 vector, matches builtin param

union FragBF { v16bf v; uint4 q[2]; };

// ---- gfx1250 async global->LDS path (guarded; falls back to sync staging) ----
#if __has_builtin(__builtin_amdgcn_global_load_async_to_lds_b128)
#define HAS_ASYNC_LDS 1
#else
#define HAS_ASYNC_LDS 0
#endif

#if HAS_ASYNC_LDS
#if __has_builtin(__builtin_amdgcn_s_wait_asynccnt)
#define WAIT_ASYNC(n) __builtin_amdgcn_s_wait_asynccnt(n)
#else
#define WAIT_ASYNC(n) asm volatile("s_wait_asynccnt %0" ::"i"(n))
#endif
__device__ __forceinline__ void async_cp16(const __bf16* g, __bf16* l) {
  __builtin_amdgcn_global_load_async_to_lds_b128(
      (__attribute__((address_space(1))) v4i_t*)(g),
      (__attribute__((address_space(3))) v4i_t*)(l), 0, 0);
}
#endif

// ---------------- precision prep ----------------
__global__ void cvt_f32_bf16(const float* __restrict__ src, __bf16* __restrict__ dst, int n) {
  int i = blockIdx.x * blockDim.x + threadIdx.x;
  if (i < n) dst[i] = (__bf16)src[i];
}

__global__ void transpose_bf16(const float* __restrict__ src, __bf16* __restrict__ dst, int n) {
  __shared__ __bf16 tile[32][33];
  int x  = blockIdx.x * 32 + threadIdx.x;
  int y0 = blockIdx.y * 32 + threadIdx.y;
#pragma unroll
  for (int j = 0; j < 32; j += 8)
    tile[threadIdx.y + j][threadIdx.x] = (__bf16)src[(size_t)(y0 + j) * n + x];
  __syncthreads();
  int x2 = blockIdx.y * 32 + threadIdx.x;
  int y2 = blockIdx.x * 32 + threadIdx.y;
#pragma unroll
  for (int j = 0; j < 32; j += 8)
    dst[(size_t)(y2 + j) * n + x2] = tile[threadIdx.x][threadIdx.y + j];
}

// ---------------- WMMA GEMM: C(MxN) = A(MxK) * Bm(NxK)^T, bf16 in / f32 acc ----------------
#define BM  128
#define BN  128
#define BK  64
#define LDT 72                 // padded LDS stride (halves); rows stay 16B aligned
#define STAGE_A (BM * LDT)     // halves per A stage
#define STAGE_B (BN * LDT)
#define GEMM_SMEM_BYTES ((2 * STAGE_A + 2 * STAGE_B) * 2)

#if HAS_ASYNC_LDS
__device__ __forceinline__ void stage_tile_async(const __bf16* Ag, const __bf16* Bg,
                                                 __bf16* As, __bf16* Bs,
                                                 int K, int kt, int tid) {
#pragma unroll
  for (int i = 0; i < 4; i++) {            // 128x64 halves per matrix, 16B per lane per op
    int c   = tid + i * 256;
    int row = c >> 3;
    int kc  = (c & 7) << 3;
    async_cp16(Ag + (size_t)row * K + kt + kc, As + row * LDT + kc);
    async_cp16(Bg + (size_t)row * K + kt + kc, Bs + row * LDT + kc);
  }
}
#endif

__device__ __forceinline__ void mma_stage(const __bf16* Asb, const __bf16* Bsb,
                                          v8f (&acc)[2][4],
                                          int waveM, int waveN, int r, int h) {
#pragma unroll
  for (int kk = 0; kk < BK; kk += 32) {
    FragBF a[2], b[4];
#pragma unroll
    for (int i = 0; i < 2; i++) {          // A 16x32 frag per ISA layout
      const __bf16* p = Asb + (waveM * 32 + i * 16 + r) * LDT + kk + h * 8;
      a[i].q[0] = *(const uint4*)(p);
      a[i].q[1] = *(const uint4*)(p + 16);
    }
#pragma unroll
    for (int j = 0; j < 4; j++) {          // B 32x16 frag per ISA layout (same k order as A)
      const __bf16* p = Bsb + (waveN * 64 + j * 16 + r) * LDT + kk + h * 8;
      b[j].q[0] = *(const uint4*)(p);
      b[j].q[1] = *(const uint4*)(p + 16);
    }
#pragma unroll
    for (int i = 0; i < 2; i++)
#pragma unroll
      for (int j = 0; j < 4; j++)
        acc[i][j] = __builtin_amdgcn_wmma_f32_16x16x32_bf16(
            false, a[i].v, false, b[j].v, (short)0, acc[i][j], false, false);
  }
}

__global__ __launch_bounds__(256)
void gemm_nt_bf16(const __bf16* __restrict__ A, const __bf16* __restrict__ Bm,
                  float* __restrict__ Cf, __bf16* __restrict__ Cbf,
                  int M, int N, int K) {
  extern __shared__ __bf16 smem[];
  __bf16* As = smem;                 // [2][STAGE_A]
  __bf16* Bs = smem + 2 * STAGE_A;   // [2][STAGE_B]
  const int tid   = threadIdx.x;
  const int lane  = tid & 31, wave = tid >> 5;
  const int waveM = wave & 3, waveN = wave >> 2;   // 4x2 wave grid -> 32x64 per wave
  const int r = lane & 15, h = lane >> 4;
  const int bm = blockIdx.y, bn = blockIdx.x;

  v8f acc[2][4] = {};

  const __bf16* Ag = A  + (size_t)bm * BM * K;
  const __bf16* Bg = Bm + (size_t)bn * BN * K;
  const int nk = K / BK;

#if HAS_ASYNC_LDS
  // double-buffered: async-copy next K slab while WMMAs consume the current one
  stage_tile_async(Ag, Bg, As, Bs, K, 0, tid);
  for (int ki = 0; ki < nk; ++ki) {
    const int s = ki & 1;
    if (ki + 1 < nk) {
      stage_tile_async(Ag, Bg, As + (s ^ 1) * STAGE_A, Bs + (s ^ 1) * STAGE_B,
                       K, (ki + 1) * BK, tid);
      WAIT_ASYNC(8);   // in-order per wave: <=8 outstanding => current stage landed
    } else {
      WAIT_ASYNC(0);
    }
    __syncthreads();
    mma_stage(As + s * STAGE_A, Bs + s * STAGE_B, acc, waveM, waveN, r, h);
    __syncthreads();   // stage s may be overwritten next iteration
  }
#else
  for (int ki = 0; ki < nk; ++ki) {
    const int kt = ki * BK;
#pragma unroll
    for (int i = 0; i < 4; i++) {
      int c   = tid + i * 256;
      int row = c >> 3;
      int kc  = (c & 7) << 3;
      *(uint4*)(As + row * LDT + kc) = *(const uint4*)(Ag + (size_t)row * K + kt + kc);
      *(uint4*)(Bs + row * LDT + kc) = *(const uint4*)(Bg + (size_t)row * K + kt + kc);
    }
    __syncthreads();
    mma_stage(As, Bs, acc, waveM, waveN, r, h);
    __syncthreads();
  }
#endif

#pragma unroll
  for (int i = 0; i < 2; i++) {
#pragma unroll
    for (int j = 0; j < 4; j++) {
      int gcol  = bn * BN + waveN * 64 + j * 16 + r;
      int growb = bm * BM + waveM * 32 + i * 16 + h * 8;
#pragma unroll
      for (int g = 0; g < 8; g++) {
        float v = acc[i][j][g];
        size_t off = (size_t)(growb + g) * N + gcol;
        if (Cf)  Cf[off]  = v;
        if (Cbf) Cbf[off] = (__bf16)v;
      }
    }
  }
}

// ---------------- exact top-OMEGA hard threshold (radix select on |v| bits) ----------------
__global__ __launch_bounds__(256)
void hardthr_kernel(const float* __restrict__ base, const float* __restrict__ add,
                    float* __restrict__ zf, __bf16* __restrict__ zbf) {
  const int row = blockIdx.x;
  const int t   = threadIdx.x;
  const float* bp = base + (size_t)row * D_DIM;
  const float* ap = add ? add + (size_t)row * D_DIM : nullptr;
  float v[8]; unsigned ab[8];
#pragma unroll
  for (int e = 0; e < 8; e++) {
    int idx = t + e * 256;
    float x = bp[idx];
    if (ap) x += ap[idx];
    v[e]  = x;
    ab[e] = __float_as_uint(fabsf(x));   // non-negative float: bit order == value order
  }
  __shared__ unsigned hist[256];
  __shared__ unsigned sk, spref;
  if (t == 0) { sk = OMEGA_K; spref = 0u; }
  __syncthreads();
#pragma unroll
  for (int lvl = 0; lvl < 4; lvl++) {
    const int shift = 24 - lvl * 8;
    hist[t] = 0u;
    __syncthreads();
    unsigned pref = spref;
    unsigned hm   = (lvl == 0) ? 0u : (0xFFFFFFFFu << (shift + 8));
#pragma unroll
    for (int e = 0; e < 8; e++)
      if ((ab[e] & hm) == pref) atomicAdd(&hist[(ab[e] >> shift) & 255u], 1u);
    __syncthreads();
    if (t == 0) {
      unsigned k = sk;
      int b = 255; unsigned accu = hist[255];
      while (accu < k && b > 0) { --b; accu += hist[b]; }
      sk    = k - (accu - hist[b]);
      spref = pref | ((unsigned)b << shift);
    }
    __syncthreads();
  }
  unsigned thr = spref;   // bit pattern of the OMEGA-th largest |v| (tie-inclusive, matches >= kth)
#pragma unroll
  for (int e = 0; e < 8; e++) {
    int idx = t + e * 256;
    float o = (ab[e] >= thr) ? v[e] : 0.0f;
    zf[(size_t)row * D_DIM + idx]  = o;
    zbf[(size_t)row * D_DIM + idx] = (__bf16)o;
  }
}

// ---------------- attention: att[p,b] = prev[p,b] . u[b]  (collapsed s_prev GEMM) ----------------
__global__ __launch_bounds__(256)
void att_kernel(const float* __restrict__ prev, const float* __restrict__ u, float* __restrict__ att) {
  const int b = blockIdx.x, t = threadIdx.x;
  __shared__ float red[256];
  float uu[8];
#pragma unroll
  for (int e = 0; e < 8; e++) uu[e] = u[(size_t)b * D_DIM + t + e * 256];
  for (int p = 0; p < P_N; p++) {
    const float* pp = prev + ((size_t)p * BATCH_N + b) * D_DIM;
    float s = 0.f;
#pragma unroll
    for (int e = 0; e < 8; e++) s += uu[e] * pp[t + e * 256];
    red[t] = s; __syncthreads();
    for (int off = 128; off > 0; off >>= 1) { if (t < off) red[t] += red[t + off]; __syncthreads(); }
    if (t == 0) att[p * BATCH_N + b] = red[0];
    __syncthreads();
  }
}

// softmax over the BATCH axis (reference: softmax(att, axis=1))
__global__ __launch_bounds__(256)
void softmax_kernel(float* __restrict__ att) {
  const int p = blockIdx.x, t = threadIdx.x;
  float* ap = att + (size_t)p * BATCH_N;
  __shared__ float red[256];
  float a[8];
  float mx = -3.4e38f;
#pragma unroll
  for (int e = 0; e < 8; e++) { a[e] = ap[t + e * 256]; mx = fmaxf(mx, a[e]); }
  red[t] = mx; __syncthreads();
  for (int off = 128; off > 0; off >>= 1) { if (t < off) red[t] = fmaxf(red[t], red[t + off]); __syncthreads(); }
  mx = red[0]; __syncthreads();
  float s = 0.f;
#pragma unroll
  for (int e = 0; e < 8; e++) { a[e] = __expf(a[e] - mx); s += a[e]; }
  red[t] = s; __syncthreads();
  for (int off = 128; off > 0; off >>= 1) { if (t < off) red[t] += red[t + off]; __syncthreads(); }
  float inv = 1.0f / red[0];
#pragma unroll
  for (int e = 0; e < 8; e++) ap[t + e * 256] = a[e] * inv;
}

__global__ __launch_bounds__(256)
void zatt_kernel(const float* __restrict__ prev, const float* __restrict__ att,
                 const float* __restrict__ lambda2,
                 float* __restrict__ zf, __bf16* __restrict__ zbf) {
  const int b = blockIdx.y;
  const int d = blockIdx.x * 256 + threadIdx.x;
  __shared__ float sa[P_N];
  if (threadIdx.x < P_N) sa[threadIdx.x] = att[threadIdx.x * BATCH_N + b];
  __syncthreads();
  float acc = 0.f;
#pragma unroll
  for (int p = 0; p < P_N; p++) {
    float pv = prev[((size_t)p * BATCH_N + b) * D_DIM + d];
    pv = fminf(fmaxf(pv, -CLAMP_V), CLAMP_V);
    acc += pv * sa[p];
  }
  acc *= lambda2[0];
  zf[(size_t)b * D_DIM + d]  = acc;
  zbf[(size_t)b * D_DIM + d] = (__bf16)acc;
}

// ---------------- spectrum + normalize ----------------
__global__ void zero_kernel(float* p, int n) {
  int i = blockIdx.x * blockDim.x + threadIdx.x;
  if (i < n) p[i] = 0.f;
}

__global__ __launch_bounds__(256)
void power_kernel(const float* __restrict__ z, float* __restrict__ mD) {
  const int w  = blockIdx.x * 256 + threadIdx.x;
  const int r0 = blockIdx.y * (BATCH_N / 64);
  float s = 0.f;
  for (int r = r0; r < r0 + BATCH_N / 64; ++r) {
    float a = z[(size_t)r * D_DIM + w];
    float c = z[(size_t)r * D_DIM + W_DIM + w];
    s += a * a + c * c;
  }
  atomicAdd(&mD[w], s);
}

__global__ __launch_bounds__(256)
void norm_kernel(const float* __restrict__ mD, float* __restrict__ out) {
  const int t = threadIdx.x;
  __shared__ float smn[256], smx[256];
  float mn = 3.4e38f, mx = -3.4e38f;
  float v[4];
#pragma unroll
  for (int e = 0; e < 4; e++) { v[e] = mD[t + e * 256]; mn = fminf(mn, v[e]); mx = fmaxf(mx, v[e]); }
  smn[t] = mn; smx[t] = mx; __syncthreads();
  for (int off = 128; off > 0; off >>= 1) {
    if (t < off) { smn[t] = fminf(smn[t], smn[t + off]); smx[t] = fmaxf(smx[t], smx[t + off]); }
    __syncthreads();
  }
  mn = smn[0]; mx = smx[0];
  float inv = 1.0f / (mx - mn + 1e-8f);
#pragma unroll
  for (int e = 0; e < 4; e++) out[t + e * 256] = (v[e] - mn) * inv;
}

// ---------------- orchestration ----------------
extern "C" void kernel_launch(void* const* d_in, const int* in_sizes, int n_in,
                              void* d_out, int out_size, void* d_ws, size_t ws_size,
                              hipStream_t stream) {
  const float* x    = (const float*)d_in[0];
  const float* prev = (const float*)d_in[1];
  const float* W_d  = (const float*)d_in[2];
  const float* S    = (const float*)d_in[3];
  const float* lam2 = (const float*)d_in[4];
  float* out = (float*)d_out;   // [0..W) = mD_norm, [W..W+BATCH*D) = z_last

  const size_t MAT = (size_t)D_DIM * D_DIM;
  const size_t ACT = (size_t)BATCH_N * D_DIM;

  __bf16* w_bf  = (__bf16*)d_ws;        // W_d           (8 MB)
  __bf16* s_bf  = w_bf  + MAT;          // S             (8 MB)
  __bf16* wT_bf = s_bf  + MAT;          // W_d^T         (8 MB)
  __bf16* actA  = wT_bf + MAT;          // x_bf -> s_t   (8 MB)
  __bf16* actB  = actA  + ACT;          // z_bf          (8 MB)
  float*  Bf    = (float*)(actB + ACT); // B = x W_d^T   (16 MB, reused every LIHT iter)
  float*  t_f   = Bf + ACT;             // GEMM f32 out  (16 MB)
  float*  z_f   = t_f + ACT;            // z f32         (16 MB)
  float*  att   = z_f + ACT;            // (P, BATCH)
  float*  mD    = att + (size_t)P_N * BATCH_N;

  dim3 cb(256);
  dim3 gg(D_DIM / 128, BATCH_N / 128);
  const size_t gsm = GEMM_SMEM_BYTES;   // 72 KB dynamic LDS (double-buffered stages)

  // precision prep (one-time per launch; bf16 operands stay L2-resident)
  cvt_f32_bf16<<<dim3((unsigned)(MAT / 256)), cb, 0, stream>>>(W_d, w_bf, (int)MAT);
  cvt_f32_bf16<<<dim3((unsigned)(MAT / 256)), cb, 0, stream>>>(S,   s_bf, (int)MAT);
  cvt_f32_bf16<<<dim3((unsigned)(ACT / 256)), cb, 0, stream>>>(x,   actA, (int)ACT);
  transpose_bf16<<<dim3(D_DIM / 32, D_DIM / 32), dim3(32, 8), 0, stream>>>(W_d, wT_bf, D_DIM);

  // B = x W_d^T ; z = thr(B)
  gemm_nt_bf16<<<gg, cb, gsm, stream>>>(actA, w_bf, Bf, nullptr, BATCH_N, D_DIM, D_DIM);
  hardthr_kernel<<<dim3(BATCH_N), cb, 0, stream>>>(Bf, nullptr, z_f, actB);
  // z = thr(B + S z)
  gemm_nt_bf16<<<gg, cb, gsm, stream>>>(actB, s_bf, t_f, nullptr, BATCH_N, D_DIM, D_DIM);
  hardthr_kernel<<<dim3(BATCH_N), cb, 0, stream>>>(Bf, t_f, z_f, actB);
  // s_t = z W_d (bf16 only), u = s_t W_d  -> att[p,b] = prev[p,b].u[b]
  gemm_nt_bf16<<<gg, cb, gsm, stream>>>(actB, wT_bf, nullptr, actA, BATCH_N, D_DIM, D_DIM);
  gemm_nt_bf16<<<gg, cb, gsm, stream>>>(actA, wT_bf, t_f, nullptr, BATCH_N, D_DIM, D_DIM);
  att_kernel<<<dim3(BATCH_N), cb, 0, stream>>>(prev, t_f, att);
  softmax_kernel<<<dim3(P_N), cb, 0, stream>>>(att);
  zatt_kernel<<<dim3(D_DIM / 256, BATCH_N), cb, 0, stream>>>(prev, att, lam2, z_f, actB);

  // LIHT iterations; final z written straight into d_out
  for (int it = 0; it < NITERS; ++it) {
    gemm_nt_bf16<<<gg, cb, gsm, stream>>>(actB, s_bf, t_f, nullptr, BATCH_N, D_DIM, D_DIM);
    float* zdst = (it == NITERS - 1) ? (out + W_DIM) : z_f;
    hardthr_kernel<<<dim3(BATCH_N), cb, 0, stream>>>(Bf, t_f, zdst, actB);
  }

  // power spectrum over batch + min/max normalize
  zero_kernel<<<dim3(W_DIM / 256), cb, 0, stream>>>(mD, W_DIM);
  power_kernel<<<dim3(W_DIM / 256, 64), cb, 0, stream>>>(out + W_DIM, mD);
  norm_kernel<<<dim3(1), cb, 0, stream>>>(mD, out);
}
